// WLNLayer_5188320494200
// MI455X (gfx1250) — matrix operand order repaired and checked by reading, not compiled, not run on hardware
//
#include <hip/hip_runtime.h>
#include <hip/hip_bf16.h>

// ---- types ------------------------------------------------------------
typedef __attribute__((ext_vector_type(16))) __bf16 v16bf;
typedef __attribute__((ext_vector_type(8)))  float  v8f;

union FragU { v16bf v; unsigned u[8]; };

__device__ __forceinline__ v8f wmma_bf16(v16bf a, v16bf b, v8f c) {
  // D = A(16x32 bf16) x B(32x16 bf16) + C(16x16 f32)
  return __builtin_amdgcn_wmma_f32_16x16x32_bf16(
      /*neg_a=*/false, a, /*neg_b=*/false, b,
      /*c_mod=*/(short)0, c, /*reuse_a=*/false, /*reuse_b=*/false);
}

// A-fragment from an LDS row-major bf16 tile.
// 16-bit A 16x32 layout (contiguous): lane<16 holds K=0..15 ; lane>=16 holds K=16..31.
// element j -> k = base_k + (base_k & 8) + j
__device__ __forceinline__ v16bf lds_a_frag(const __bf16* buf, int row_stride,
                                            int m, int base_k) {
  FragU fr;
#pragma unroll
  for (int p = 0; p < 8; ++p) {
    int j = 2 * p;
    int k = base_k + (base_k & 8) + j;
    fr.u[p] = *reinterpret_cast<const unsigned*>(buf + m * row_stride + k);
  }
  return fr.v;
}

// ---- weight pre-pack: f32 -> bf16 in B-fragment order -----------------
// B 32x16 layout: lanes0-15 K=0..15 (elem j -> k=j), lanes16-31 K=16..31.
// w1 packed frags: ((t*8+kb)*8+nt), w2: ((t*4+kb)*8+nt), sw: (kb*8+nt)
// Each frag = 32 lanes * 16 bf16 = 512 elems = 1KB.
#define W1P_ELEMS 131072   // 4*8*8*512
#define W2P_ELEMS 65536    // 4*4*8*512
#define SWP_ELEMS 16384    // 4*8*512
#define PACK_TOTAL (W1P_ELEMS + W2P_ELEMS + SWP_ELEMS)  // 212992

__global__ void wln_pack_weights(const float* __restrict__ w1,
                                 const float* __restrict__ w2,
                                 const float* __restrict__ sw,
                                 unsigned short* __restrict__ out) {
  int i = blockIdx.x * 256 + threadIdx.x;
  float val;
  if (i < W1P_ELEMS) {
    int frag = i >> 9, r = i & 511;
    int lane = r >> 4, j = r & 15;
    int nt = frag & 7, kb = (frag >> 3) & 7, t = frag >> 6;
    int k = kb * 32 + ((lane >> 4) << 4) + j;
    int n = (nt << 4) + (lane & 15);
    val = w1[(t * 256 + k) * 128 + n];
  } else if (i < W1P_ELEMS + W2P_ELEMS) {
    int i2 = i - W1P_ELEMS;
    int frag = i2 >> 9, r = i2 & 511;
    int lane = r >> 4, j = r & 15;
    int nt = frag & 7, kb = (frag >> 3) & 3, t = frag >> 5;
    int k = kb * 32 + ((lane >> 4) << 4) + j;
    int n = (nt << 4) + (lane & 15);
    val = w2[(t * 128 + k) * 128 + n];
  } else {
    int i3 = i - (W1P_ELEMS + W2P_ELEMS);
    int frag = i3 >> 9, r = i3 & 511;
    int lane = r >> 4, j = r & 15;
    int nt = frag & 7, kb = frag >> 3;
    int k = kb * 32 + ((lane >> 4) << 4) + j;
    int n = (nt << 4) + (lane & 15);
    val = sw[k * 128 + n];
  }
  union { __bf16 b; unsigned short s; } cv;
  cv.b = (__bf16)val;
  out[i] = cv.s;
}

// ---- self-loop GEMM: out = x @ self_w + self_b (initializes d_out) ----
__global__ void wln_self_gemm(const float* __restrict__ x,
                              const float* __restrict__ sb,
                              const v16bf* __restrict__ swp,
                              float* __restrict__ out, int N) {
  __shared__ __bf16 xa[16 * 128];
  int tid = threadIdx.x;
  int base = blockIdx.x * 16;
#pragma unroll
  for (int it = 0; it < 8; ++it) {
    int idx = it * 256 + tid;
    int e = idx >> 7, f = idx & 127;
    int node = base + e; if (node >= N) node = N - 1;
    xa[idx] = (__bf16)x[(size_t)node * 128 + f];
  }
  __syncthreads();
  int lane = tid & 31, w = tid >> 5;
  int hi = lane >> 4, l15 = lane & 15;
  v8f acc = {};
#pragma unroll
  for (int kb = 0; kb < 4; ++kb) {
    v16bf a = lds_a_frag(xa, 128, l15, kb * 32 + hi * 8);
    v16bf b = swp[(kb * 8 + w) * 32 + lane];
    acc = wmma_bf16(a, b, acc);
  }
  int c = w * 16 + l15;
  float bias = sb[c];
#pragma unroll
  for (int r = 0; r < 8; ++r) {
    int m = base + r + hi * 8;
    if (m < N) out[(size_t)m * 128 + c] = acc[r] + bias;
  }
}

// ---- edge MLP + scatter: 64 edges per block, 8 waves -------------------
// Each wave owns one 16-column tile and 4 row tiles (4x B-fragment reuse).
#define TILE_E 64

__global__ void wln_edge_mlp(const float* __restrict__ x,
                             const int* __restrict__ eidx,
                             const int* __restrict__ eattr,
                             const float* __restrict__ b1,
                             const float* __restrict__ b2,
                             const v16bf* __restrict__ w1p,
                             const v16bf* __restrict__ w2p,
                             float* __restrict__ out, int E) {
  __shared__ int row_s[TILE_E], col_s[TILE_E], attr_s[TILE_E];
  __shared__ __bf16 ef[TILE_E * 256];   // concat(x[row], x[col]) bf16 (32KB)
  __shared__ __bf16 hb[TILE_E * 128];   // layer-1 activations (16KB)
  int tid = threadIdx.x;
  int e0 = blockIdx.x * TILE_E;
  if (tid < TILE_E) {
    int e = e0 + tid; if (e >= E) e = E - 1;
    row_s[tid] = eidx[e];
  } else if (tid < 2 * TILE_E) {
    int e = e0 + (tid - TILE_E); if (e >= E) e = E - 1;
    col_s[tid - TILE_E] = eidx[E + e];
  } else if (tid < 3 * TILE_E) {
    int e = e0 + (tid - 2 * TILE_E); if (e >= E) e = E - 1;
    attr_s[tid - 2 * TILE_E] = eattr[e];
  }
  __syncthreads();
  int f = tid;  // 0..255 -> feature index of concat vector
#pragma unroll 8
  for (int e = 0; e < TILE_E; ++e) {
    float v = (f < 128) ? x[(size_t)row_s[e] * 128 + f]
                        : x[(size_t)col_s[e] * 128 + (f - 128)];
    ef[e * 256 + f] = (__bf16)v;
  }
  __syncthreads();

  int lane = tid & 31, w = tid >> 5;
  int hi = lane >> 4, l15 = lane & 15;
  int c = w * 16 + l15;             // output column owned by this lane
  v8f msg[4] = {};                  // one 16x16 tile per row-block
  for (int t = 0; t < 4; ++t) {
    // layer 1: (64x256) @ (256x128); each B fragment feeds 4 WMMAs
    v8f acc[4] = {};
#pragma unroll
    for (int kb = 0; kb < 8; ++kb) {
      v16bf bm = w1p[((t * 8 + kb) * 8 + w) * 32 + lane];
#pragma unroll
      for (int mt = 0; mt < 4; ++mt) {
        v16bf a = lds_a_frag(ef, 256, mt * 16 + l15, kb * 32 + hi * 8);
        acc[mt] = wmma_bf16(a, bm, acc[mt]);
      }
    }
    float bias1 = b1[t * 128 + c];
#pragma unroll
    for (int mt = 0; mt < 4; ++mt) {
#pragma unroll
      for (int r = 0; r < 8; ++r) {
        float h = acc[mt][r] + bias1;
        h = h > 0.f ? h : 0.f;                              // ReLU
        hb[(mt * 16 + r + hi * 8) * 128 + c] = (__bf16)h;   // D-layout -> LDS
      }
    }
    __syncthreads();
    // layer 2: (64x128) @ (128x128)
    v8f acc2[4] = {};
#pragma unroll
    for (int kb = 0; kb < 4; ++kb) {
      v16bf bm = w2p[((t * 4 + kb) * 8 + w) * 32 + lane];
#pragma unroll
      for (int mt = 0; mt < 4; ++mt) {
        v16bf a = lds_a_frag(hb, 128, mt * 16 + l15, kb * 32 + hi * 8);
        acc2[mt] = wmma_bf16(a, bm, acc2[mt]);
      }
    }
    float bias2 = b2[t * 128 + c];
#pragma unroll
    for (int mt = 0; mt < 4; ++mt) {
#pragma unroll
      for (int r = 0; r < 8; ++r) {
        int m = mt * 16 + r + hi * 8;                 // local edge index
        msg[mt][r] += (attr_s[m] == t) ? (acc2[mt][r] + bias2) : 0.f;
      }
    }
    __syncthreads();  // protect hb before next type overwrites it
  }
  // scatter-add into destination nodes (L2 atomics)
#pragma unroll
  for (int mt = 0; mt < 4; ++mt) {
#pragma unroll
    for (int r = 0; r < 8; ++r) {
      int m = mt * 16 + r + hi * 8;
      int e = e0 + m;
      if (e < E) atomicAdd(&out[(size_t)col_s[m] * 128 + c], msg[mt][r]);
    }
  }
}

// ---- ReLU + LayerNorm, one wave per node ------------------------------
__global__ void wln_relu_ln(float* __restrict__ out,
                            const float* __restrict__ g,
                            const float* __restrict__ bt, int N) {
  int tid = threadIdx.x;
  int lane = tid & 31, w = tid >> 5;
  int n = blockIdx.x * 8 + w;
  if (n >= N) return;  // uniform per wave
  float4* p = reinterpret_cast<float4*>(out + (size_t)n * 128);
  float4 v = p[lane];
  v.x = fmaxf(v.x, 0.f); v.y = fmaxf(v.y, 0.f);
  v.z = fmaxf(v.z, 0.f); v.w = fmaxf(v.w, 0.f);
  float s = v.x + v.y + v.z + v.w;
  float q = v.x * v.x + v.y * v.y + v.z * v.z + v.w * v.w;
#pragma unroll
  for (int off = 16; off > 0; off >>= 1) {
    s += __shfl_xor(s, off, 32);
    q += __shfl_xor(q, off, 32);
  }
  float mean = s * (1.f / 128.f);
  float var  = q * (1.f / 128.f) - mean * mean;
  float rstd = rsqrtf(var + 1e-5f);
  float4 gg = reinterpret_cast<const float4*>(g)[lane];
  float4 bb = reinterpret_cast<const float4*>(bt)[lane];
  float4 y;
  y.x = (v.x - mean) * rstd * gg.x + bb.x;
  y.y = (v.y - mean) * rstd * gg.y + bb.y;
  y.z = (v.z - mean) * rstd * gg.z + bb.z;
  y.w = (v.w - mean) * rstd * gg.w + bb.w;
  p[lane] = y;
}

// ---- launcher ----------------------------------------------------------
extern "C" void kernel_launch(void* const* d_in, const int* in_sizes, int n_in,
                              void* d_out, int out_size, void* d_ws, size_t ws_size,
                              hipStream_t stream) {
  const float* x    = (const float*)d_in[0];
  const int*   eidx = (const int*)  d_in[1];
  const int*   eatt = (const int*)  d_in[2];
  const float* w1   = (const float*)d_in[3];
  const float* b1   = (const float*)d_in[4];
  const float* w2   = (const float*)d_in[5];
  const float* b2   = (const float*)d_in[6];
  const float* sw   = (const float*)d_in[7];
  const float* sb   = (const float*)d_in[8];
  const float* g    = (const float*)d_in[9];
  const float* bt   = (const float*)d_in[10];
  float* out = (float*)d_out;
  int N = in_sizes[0] / 128;
  int E = in_sizes[2];

  unsigned short* wsp = (unsigned short*)d_ws;
  const v16bf* w1p = (const v16bf*)d_ws;              // 8192 frag-units
  const v16bf* w2p = w1p + (W1P_ELEMS * 2 / 32);      // +8192
  const v16bf* swp = w2p + (W2P_ELEMS * 2 / 32);      // +4096

  wln_pack_weights<<<PACK_TOTAL / 256, 256, 0, stream>>>(w1, w2, sw, wsp);
  wln_self_gemm<<<(N + 15) / 16, 256, 0, stream>>>(x, sb, swp, out, N);
  wln_edge_mlp<<<(E + TILE_E - 1) / TILE_E, 256, 0, stream>>>(
      x, eidx, eatt, b1, b2, w1p, w2p, out, E);
  wln_relu_ln<<<(N + 7) / 8, 256, 0, stream>>>(out, g, bt, N);
}
